// CrossModalAttention_89988154786234
// MI455X (gfx1250) — compile-verified, matrix operand order checked by reading, not run on hardware
//
#include <hip/hip_runtime.h>
#include <hip/hip_bf16.h>

typedef _Float16 f16;
typedef __attribute__((ext_vector_type(8)))  _Float16 v8h;
typedef __attribute__((ext_vector_type(16))) _Float16 v16h;
typedef __attribute__((ext_vector_type(8)))  float    v8f;

#define NHEAD 8
#define HD    32
#define CD    256
#define NPIX  3136   // 56*56
#define BATCH 2

// ---------------------------------------------------------------------------
// Fragment loader for 16x16x32 f16 WMMA (wave32).
// A-matrix (16 rows x 32 K) and B-matrix (32 K x 16 cols) use the mirrored
// CDNA5 striping: lane l<16 -> row/col (l), K = 0..7 & 16..23;
// lane l>=16 -> row/col (l-16), K = 8..15 & 24..31.
// element (r, k) lives at base + r*ld + k (K contiguous).
// ---------------------------------------------------------------------------
__device__ __forceinline__ v16h load_frag16(const f16* base, int ld) {
  const int l = threadIdx.x & 31;
  const int r = l & 15;
  const int s = (l >> 4) << 3;          // 0 or 8
  const f16* p = base + (size_t)r * ld + s;
  v8h lo = *(const v8h*)(p);
  v8h hi = *(const v8h*)(p + 16);
  v16h out;
#pragma unroll
  for (int i = 0; i < 8; ++i) { out[i] = lo[i]; out[i + 8] = hi[i]; }
  return out;
}

// ---------------------------------------------------------------------------
// f32 -> f16 elementwise convert (weights)
// ---------------------------------------------------------------------------
__global__ void k_cvt_f16(const float* __restrict__ src, f16* __restrict__ dst, int n) {
  int i = blockIdx.x * blockDim.x + threadIdx.x;
  if (i < n) dst[i] = (f16)src[i];
}

// ---------------------------------------------------------------------------
// x (B,C,N) f32 -> xT (B,N,C) f16, LDS-tiled 32x32 transpose (coalesced both
// directions; 33-stride padding kills LDS bank conflicts).
// grid: B * (C/32) * (N/32) blocks of 256 threads; each thread moves 4 elems.
// ---------------------------------------------------------------------------
__global__ void k_transpose_cvt(const float* __restrict__ x, f16* __restrict__ xT) {
  __shared__ f16 t[32][33];
  const int bx = blockIdx.x;
  const int n0 = (bx % (NPIX / 32)) * 32;
  const int c0 = ((bx / (NPIX / 32)) % (CD / 32)) * 32;
  const int b  = bx / ((NPIX / 32) * (CD / 32));
  const int tx = threadIdx.x & 31;
  const int ty0 = threadIdx.x >> 5;        // 0..7
#pragma unroll
  for (int i = 0; i < 4; ++i) {
    int ty = ty0 + i * 8;
    t[ty][tx] = (f16)x[(size_t)(b * CD + c0 + ty) * NPIX + n0 + tx];
  }
  __syncthreads();
#pragma unroll
  for (int i = 0; i < 4; ++i) {
    int ty = ty0 + i * 8;
    xT[(size_t)(b * NPIX + n0 + ty) * CD + c0 + tx] = t[tx][ty];
  }
}

// ---------------------------------------------------------------------------
// Tiled WMMA GEMM:  out(M=256, N=3136 per batch) = A(256,K) @ B(b,:,K)^T
//   A : (256, K) f16 row-major (weights)
//   Bm: (BATCH, NPIX, K) f16   (activations, channel-contiguous)
// One 32(M) x 64(N) tile per wave: 2 A frags x 4 B frags -> 8 WMMAs per
// k-step (0.75 fragment loads per WMMA); 8 waves per block.
// mode 0: f16 store transposed (B,N,256), +bias
// mode 1: f16 store natural    (B,256,N), +bias
// mode 2: f16 store transposed (B,N,256), +bias, BN(eval)+ReLU
// mode 3: f32 store natural    (B,256,N), +bias  (final output)
// ---------------------------------------------------------------------------
__global__ void k_gemm(const f16* __restrict__ A,
                       const f16* __restrict__ Bm,
                       const float* __restrict__ bias,
                       const float* __restrict__ gamma,
                       const float* __restrict__ beta,
                       f16* __restrict__ outh,
                       float* __restrict__ outf,
                       int K, int mode)
{
  const int wave = threadIdx.x >> 5;
  const int tid  = blockIdx.x * 8 + wave;
  const int NT   = NPIX / 64;                       // 49 strips in N
  const int nt = tid % NT;
  const int mt = (tid / NT) % (CD / 32);
  const int b  = tid / (NT * (CD / 32));
  const int m0 = mt * 32, n0 = nt * 64;

  v8f acc[2][4] = {};
  const f16* bbase = Bm + (size_t)(b * NPIX + n0) * K;
  const f16* abase = A + (size_t)m0 * K;
  for (int k0 = 0; k0 < K; k0 += 32) {
    v16h af0 = load_frag16(abase + k0, K);
    v16h af1 = load_frag16(abase + (size_t)16 * K + k0, K);
#pragma unroll
    for (int j = 0; j < 4; ++j) {
      v16h bf = load_frag16(bbase + (size_t)(j * 16) * K + k0, K);
      acc[0][j] = __builtin_amdgcn_wmma_f32_16x16x32_f16(false, af0, false, bf,
                                                         (short)0, acc[0][j], false, false);
      acc[1][j] = __builtin_amdgcn_wmma_f32_16x16x32_f16(false, af1, false, bf,
                                                         (short)0, acc[1][j], false, false);
    }
  }

  const int l  = threadIdx.x & 31;
  const int lc = l & 15;
  const int rsh = (l >> 4) << 3;

#pragma unroll
  for (int i = 0; i < 2; ++i) {
    const int mb = m0 + i * 16 + rsh;
#pragma unroll
    for (int j = 0; j < 4; ++j) {
      const int n = n0 + j * 16 + lc;
      if (mode == 0) {
        v8h pk;
#pragma unroll
        for (int v = 0; v < 8; ++v) pk[v] = (f16)(acc[i][j][v] + bias[mb + v]);
        *(v8h*)(outh + (size_t)(b * NPIX + n) * CD + mb) = pk;
      } else if (mode == 1) {
#pragma unroll
        for (int v = 0; v < 8; ++v)
          outh[(size_t)(b * CD + mb + v) * NPIX + n] = (f16)(acc[i][j][v] + bias[mb + v]);
      } else if (mode == 2) {
        const float invs = rsqrtf(1.0f + 1e-5f);
        v8h pk;
#pragma unroll
        for (int v = 0; v < 8; ++v) {
          float h = (acc[i][j][v] + bias[mb + v]) * invs * gamma[mb + v] + beta[mb + v];
          pk[v] = (f16)fmaxf(h, 0.0f);
        }
        *(v8h*)(outh + (size_t)(b * NPIX + n) * CD + mb) = pk;
      } else {
#pragma unroll
        for (int v = 0; v < 8; ++v)
          outf[(size_t)(b * CD + mb + v) * NPIX + n] = acc[i][j][v] + bias[mb + v];
      }
    }
  }
}

// ---------------------------------------------------------------------------
// Flash attention per (batch, head, 16-query tile).  One wave per block.
//   Qb,Kb : (B, N, 256) f16 ; head h occupies channels [h*32, h*32+32)
//   Vb    : (B, 256, N) f16 ; key-contiguous columns for the PV GEMM
//   out   : fused (B, N, 512) f16, written at channel offset coff + h*32
// Online softmax, 32 keys per iteration:
//   2 WMMAs for scores, LDS round-trip to A-layout, 2 WMMAs into O.
// ---------------------------------------------------------------------------
__global__ void k_flash(const f16* __restrict__ Qb,
                        const f16* __restrict__ Kb,
                        const f16* __restrict__ Vb,
                        f16* __restrict__ fused,
                        int coff)
{
  __shared__ __align__(16) f16 pbuf[16 * 32];

  const int blk = blockIdx.x;
  const int qt  = blk % (NPIX / 16);
  const int h   = (blk / (NPIX / 16)) % NHEAD;
  const int b   = blk / ((NPIX / 16) * NHEAD);
  const int q0  = qt * 16;
  const int l   = threadIdx.x & 31;
  const int lc  = l & 15;               // column / N index within tile
  const int rsh = (l >> 4) << 3;        // row offset 0 or 8 for C-tiles
  const float sc = 0.17677669529663687f;  // 1/sqrt(32)

  v16h qf = load_frag16(Qb + (size_t)(b * NPIX + q0) * CD + h * HD, CD);

  v8f o0 = {}, o1 = {};
  float mrow[8], lrow[8];
#pragma unroll
  for (int v = 0; v < 8; ++v) { mrow[v] = -3.0e38f; lrow[v] = 0.0f; }

  for (int m0 = 0; m0 < NPIX; m0 += 32) {
    // ---- scores: S(16x32) = Q(16x32) @ K^T, one WMMA per 16-key half ----
    const f16* kbase = Kb + (size_t)(b * NPIX + m0) * CD + h * HD;
    v16h kf0 = load_frag16(kbase, CD);
    v16h kf1 = load_frag16(kbase + (size_t)16 * CD, CD);
    v8f z = {};
    v8f s0 = __builtin_amdgcn_wmma_f32_16x16x32_f16(false, qf, false, kf0,
                                                    (short)0, z, false, false);
    v8f s1 = __builtin_amdgcn_wmma_f32_16x16x32_f16(false, qf, false, kf1,
                                                    (short)0, z, false, false);

    // ---- online softmax (per-row stats across the 16-lane row group) ----
    float mnew[8], alpha[8];
#pragma unroll
    for (int v = 0; v < 8; ++v) {
      s0[v] *= sc; s1[v] *= sc;
      float t = fmaxf(s0[v], s1[v]);
#pragma unroll
      for (int mk = 8; mk >= 1; mk >>= 1) t = fmaxf(t, __shfl_xor(t, mk, 16));
      mnew[v]  = fmaxf(mrow[v], t);
      alpha[v] = __expf(mrow[v] - mnew[v]);
      float p0 = __expf(s0[v] - mnew[v]);
      float p1 = __expf(s1[v] - mnew[v]);
      s0[v] = p0; s1[v] = p1;
      float ps = p0 + p1;
#pragma unroll
      for (int mk = 8; mk >= 1; mk >>= 1) ps += __shfl_xor(ps, mk, 16);
      lrow[v] = lrow[v] * alpha[v] + ps;
      o0[v] *= alpha[v];
      o1[v] *= alpha[v];
      mrow[v] = mnew[v];
    }

    // ---- P: C-layout -> A-layout via LDS ----
#pragma unroll
    for (int v = 0; v < 8; ++v) {
      int q = v + rsh;
      pbuf[q * 32 + lc]      = (f16)s0[v];
      pbuf[q * 32 + lc + 16] = (f16)s1[v];
    }
    __syncthreads();
    v16h pa = load_frag16(pbuf, 32);
    __syncthreads();

    // ---- O += P(16x32) @ V(32x32), two WMMAs (d halves) ----
    const f16* vbase = Vb + (size_t)(b * CD + h * HD) * NPIX + m0;
    v16h vf0 = load_frag16(vbase, NPIX);
    v16h vf1 = load_frag16(vbase + (size_t)16 * NPIX, NPIX);
    o0 = __builtin_amdgcn_wmma_f32_16x16x32_f16(false, pa, false, vf0,
                                                (short)0, o0, false, false);
    o1 = __builtin_amdgcn_wmma_f32_16x16x32_f16(false, pa, false, vf1,
                                                (short)0, o1, false, false);
  }

  // ---- normalize and store into fused (B,N,512) buffer ----
#pragma unroll
  for (int v = 0; v < 8; ++v) {
    int q = v + rsh;
    float inv = 1.0f / lrow[v];
    f16* op = fused + (size_t)(b * NPIX + q0 + q) * (2 * CD) + coff + h * HD;
    op[lc]      = (f16)(o0[v] * inv);
    op[lc + 16] = (f16)(o1[v] * inv);
  }
}

// ---------------------------------------------------------------------------
// attention map: att[b] = (1/NH) * sum_c mean_n(Q[b,n,c]) * mean_n(K[b,n,c])
// broadcast over (H,W).
// ---------------------------------------------------------------------------
__global__ void k_amap(const f16* __restrict__ Qb, const f16* __restrict__ Kb,
                       float* __restrict__ amap)
{
  __shared__ float red[256];
  const int b = blockIdx.x;
  const int c = threadIdx.x;
  float sq = 0.f, sk = 0.f;
  for (int n = 0; n < NPIX; ++n) {
    sq += (float)Qb[(size_t)(b * NPIX + n) * CD + c];
    sk += (float)Kb[(size_t)(b * NPIX + n) * CD + c];
  }
  red[c] = (sq / NPIX) * (sk / NPIX);
  __syncthreads();
  for (int s = 128; s > 0; s >>= 1) {
    if (c < s) red[c] += red[c + s];
    __syncthreads();
  }
  float att = red[0] / NHEAD;
  for (int n = c; n < NPIX; n += 256)
    amap[b * NPIX + n] = att;
}

// ---------------------------------------------------------------------------
extern "C" void kernel_launch(void* const* d_in, const int* in_sizes, int n_in,
                              void* d_out, int out_size, void* d_ws, size_t ws_size,
                              hipStream_t stream)
{
  const float* rgb   = (const float*)d_in[0];
  const float* dep   = (const float*)d_in[1];
  const float* Wq_rd = (const float*)d_in[2];  const float* bq_rd = (const float*)d_in[3];
  const float* Wk_d  = (const float*)d_in[4];  const float* bk_d  = (const float*)d_in[5];
  const float* Wv_d  = (const float*)d_in[6];  const float* bv_d  = (const float*)d_in[7];
  const float* Wq_dr = (const float*)d_in[8];  const float* bq_dr = (const float*)d_in[9];
  const float* Wk_r  = (const float*)d_in[10]; const float* bk_r  = (const float*)d_in[11];
  const float* Wv_r  = (const float*)d_in[12]; const float* bv_r  = (const float*)d_in[13];
  const float* Wf1   = (const float*)d_in[14]; const float* bf1   = (const float*)d_in[15];
  const float* gamma = (const float*)d_in[16]; const float* beta  = (const float*)d_in[17];
  const float* Wf2   = (const float*)d_in[18]; const float* bf2   = (const float*)d_in[19];

  // ---- workspace carve-up (all f16, all offsets 16B aligned) ----
  const size_t XEL = (size_t)BATCH * NPIX * CD;       // 1,605,632 elements
  f16* w = (f16*)d_ws;
  size_t off = 0;
  auto alloc = [&](size_t nelem) { f16* p = w + off; off += nelem; return p; };
  f16* xTr  = alloc(XEL);
  f16* xTd  = alloc(XEL);
  f16* wqrd = alloc(CD * CD);
  f16* wkd  = alloc(CD * CD);
  f16* wvd  = alloc(CD * CD);
  f16* wqdr = alloc(CD * CD);
  f16* wkr  = alloc(CD * CD);
  f16* wvr  = alloc(CD * CD);
  f16* wf1h = alloc(CD * 2 * CD);
  f16* wf2h = alloc(CD * CD);
  f16* qrd  = alloc(XEL);
  f16* kd   = alloc(XEL);
  f16* vd   = alloc(XEL);
  f16* qdr  = alloc(XEL);
  f16* kr   = alloc(XEL);
  f16* vr   = alloc(XEL);
  f16* fusedb = alloc(2 * XEL);                       // (B, N, 512)
  f16* hbuf   = alloc(XEL);                           // (B, N, 256)

  // 1) activations -> (B,N,C) f16 (LDS-tiled transpose)
  const int TB = BATCH * (CD / 32) * (NPIX / 32);     // 1568 blocks
  k_transpose_cvt<<<TB, 256, 0, stream>>>(rgb, xTr);
  k_transpose_cvt<<<TB, 256, 0, stream>>>(dep, xTd);

  // 2) weights -> f16
  k_cvt_f16<<<(CD * CD) / 256, 256, 0, stream>>>(Wq_rd, wqrd, CD * CD);
  k_cvt_f16<<<(CD * CD) / 256, 256, 0, stream>>>(Wk_d,  wkd,  CD * CD);
  k_cvt_f16<<<(CD * CD) / 256, 256, 0, stream>>>(Wv_d,  wvd,  CD * CD);
  k_cvt_f16<<<(CD * CD) / 256, 256, 0, stream>>>(Wq_dr, wqdr, CD * CD);
  k_cvt_f16<<<(CD * CD) / 256, 256, 0, stream>>>(Wk_r,  wkr,  CD * CD);
  k_cvt_f16<<<(CD * CD) / 256, 256, 0, stream>>>(Wv_r,  wvr,  CD * CD);
  k_cvt_f16<<<(CD * 2 * CD) / 256, 256, 0, stream>>>(Wf1, wf1h, CD * 2 * CD);
  k_cvt_f16<<<(CD * CD) / 256, 256, 0, stream>>>(Wf2,  wf2h, CD * CD);

  // 3) six 1x1-conv projections via WMMA GEMM (32x64 tile per wave)
  const int GB = (BATCH * (CD / 32) * (NPIX / 64)) / 8;   // 98 blocks
  k_gemm<<<GB, 256, 0, stream>>>(wqrd, xTr, bq_rd, nullptr, nullptr, qrd, nullptr, CD, 0);
  k_gemm<<<GB, 256, 0, stream>>>(wkd,  xTd, bk_d,  nullptr, nullptr, kd,  nullptr, CD, 0);
  k_gemm<<<GB, 256, 0, stream>>>(wvd,  xTd, bv_d,  nullptr, nullptr, vd,  nullptr, CD, 1);
  k_gemm<<<GB, 256, 0, stream>>>(wqdr, xTd, bq_dr, nullptr, nullptr, qdr, nullptr, CD, 0);
  k_gemm<<<GB, 256, 0, stream>>>(wkr,  xTr, bk_r,  nullptr, nullptr, kr,  nullptr, CD, 0);
  k_gemm<<<GB, 256, 0, stream>>>(wvr,  xTr, bv_r,  nullptr, nullptr, vr,  nullptr, CD, 1);

  // 4) cross-modal flash attention (writes concat buffer directly)
  const int FB = BATCH * NHEAD * (NPIX / 16);             // 3136 blocks, 1 wave each
  k_flash<<<FB, 32, 0, stream>>>(qrd, kd, vd, fusedb, 0);     // rgb_attended
  k_flash<<<FB, 32, 0, stream>>>(qdr, kr, vr, fusedb, CD);    // depth_attended

  // 5) fusion: conv(512->256) + BN + ReLU, then conv(256->256) -> f32 out
  k_gemm<<<GB, 256, 0, stream>>>(wf1h, fusedb, bf1, gamma, beta, hbuf, nullptr, 2 * CD, 2);
  k_gemm<<<GB, 256, 0, stream>>>(wf2h, hbuf,   bf2, nullptr, nullptr, nullptr,
                                 (float*)d_out, CD, 3);

  // 6) attention map appended after fused_features
  k_amap<<<BATCH, 256, 0, stream>>>(qrd, kd, (float*)d_out + XEL);
}
